// HigherOrderGATLayer_61942018342919
// MI455X (gfx1250) — compile-verified, matrix operand order checked by reading, not run on hardware
//
#include <hip/hip_runtime.h>
#include <hip/hip_bf16.h>

// ---------------------------------------------------------------------------
// HigherOrderGATLayer on MI455X (gfx1250)
//
// Cost model: 3x { GEMM 50000x128x128 (WMMA f16, ~1.6 GFLOP, trivial) +
// 850K-edge segment-softmax scatter (~0.9 GB of gather+atomic traffic/hop,
// L2-resident out/denom/m) }  => memory/atomic bound, ~120us floor at
// 23.3 TB/s. GEMM uses v_wmma_f32_16x16x32_f16; edge phase uses native u32
// atomic-max (monotone float keys) and native f32 global atomic adds.
// ---------------------------------------------------------------------------

typedef __attribute__((ext_vector_type(16))) _Float16 v16h;
typedef __attribute__((ext_vector_type(8)))  _Float16 v8h;
typedef __attribute__((ext_vector_type(8)))  float    v8f;

#define CH 128
#define NEG_SLOPE 0.2f

// Monotone order-preserving float -> uint key (for native u32 atomic max).
static __device__ __forceinline__ unsigned enc_f32(float f) {
  unsigned u = __float_as_uint(f);
  return (u & 0x80000000u) ? ~u : (u | 0x80000000u);
}
static __device__ __forceinline__ float dec_f32(unsigned k) {
  return (k & 0x80000000u) ? __uint_as_float(k & 0x7fffffffu)
                           : __uint_as_float(~k);
}

static __device__ __forceinline__ void atomic_add_f32(float* p, float v) {
  __hip_atomic_fetch_add(p, v, __ATOMIC_RELAXED, __HIP_MEMORY_SCOPE_AGENT);
}

// --------------------------- one-time kernels ------------------------------

__global__ void k_cvt_x(const float* __restrict__ x, _Float16* __restrict__ x16,
                        int total) {
  int i = blockIdx.x * blockDim.x + threadIdx.x;
  if (i < total) x16[i] = (_Float16)x[i];
}

// out = s0*bias0 + s1*bias1 + s2*bias2  (scales folded; hop kernels then
// atomically accumulate scale_h * gat_h on top).
__global__ void k_init_out(float* __restrict__ out,
                           const float* __restrict__ b0,
                           const float* __restrict__ b1,
                           const float* __restrict__ b2,
                           float s0, float s1, float s2, int total) {
  int i = blockIdx.x * blockDim.x + threadIdx.x;
  if (i >= total) return;
  int c = i & (CH - 1);
  out[i] = s0 * b0[c] + s1 * b1[c] + s2 * b2[c];
}

// ---------------------------- per-hop kernels ------------------------------

// W (K-major, [k][n]) -> Wt16 (n-major, [n][k]) fp16, so B fragments are two
// contiguous 16B loads per lane.
__global__ void k_cvt_wt(const float* __restrict__ W, _Float16* __restrict__ wt16) {
  int i = blockIdx.x * blockDim.x + threadIdx.x;
  if (i >= CH * CH) return;
  int nn = i >> 7, kk = i & (CH - 1);
  wt16[nn * CH + kk] = (_Float16)W[kk * CH + nn];
}

// xp[n][128] = x16[n][128] @ W[128][128]. One wave per 16-row tile; 4 A frags
// held in regs; 8 column tiles x 4 K-steps of v_wmma_f32_16x16x32_f16.
__global__ __launch_bounds__(128)
void k_gemm_wmma(const _Float16* __restrict__ x16,
                 const _Float16* __restrict__ wt16,
                 float* __restrict__ xp, int n) {
  const int wave = blockIdx.x * 4 + (threadIdx.x >> 5);
  const int lane = threadIdx.x & 31;
  const int row0 = wave * 16;
  if (row0 >= n) return;
  const int hs = lane >> 4;   // half-wave select (K sub-block)
  const int lr = lane & 15;   // row (A) / col (B) within tile
  const bool full_tile = (row0 + 16 <= n);

  int arow = row0 + lr;
  if (arow >= n) arow = n - 1;                 // clamp (dup rows, stores guarded)
  const _Float16* ap = x16 + (size_t)arow * CH;

  v16h afrag[4];
#pragma unroll
  for (int k = 0; k < 4; ++k) {
    const int kb = k * 32 + hs * 8;
    v8h lo = *(const v8h*)(ap + kb);
    v8h hi = *(const v8h*)(ap + kb + 16);
    afrag[k] = __builtin_shufflevector(lo, hi, 0, 1, 2, 3, 4, 5, 6, 7,
                                       8, 9, 10, 11, 12, 13, 14, 15);
  }

#pragma unroll
  for (int t = 0; t < 8; ++t) {
    const _Float16* bp = wt16 + (size_t)(t * 16 + lr) * CH;
    v8f acc = {};
#pragma unroll
    for (int k = 0; k < 4; ++k) {
      const int kb = k * 32 + hs * 8;
      v8h lo = *(const v8h*)(bp + kb);
      v8h hi = *(const v8h*)(bp + kb + 16);
      v16h bfrag = __builtin_shufflevector(lo, hi, 0, 1, 2, 3, 4, 5, 6, 7,
                                           8, 9, 10, 11, 12, 13, 14, 15);
      acc = __builtin_amdgcn_wmma_f32_16x16x32_f16(
          /*neg_a=*/false, afrag[k], /*neg_b=*/false, bfrag,
          /*c_mod=*/(short)0, acc, /*reuse_a=*/false, /*reuse_b=*/false);
    }
    // C/D layout: VGPR r holds row (row0 + r) for lanes 0-15, (row0 + r + 8)
    // for lanes 16-31; column = t*16 + lr. One base pointer, r*CH immediate
    // offsets -> branch-free s_clause of 8 global_store_b32 on the fast path.
    float* op = xp + (size_t)(row0 + 8 * hs) * CH + t * 16 + lr;
    if (full_tile) {
#pragma unroll
      for (int r = 0; r < 8; ++r) op[r * CH] = acc[r];
    } else {
#pragma unroll
      for (int r = 0; r < 8; ++r)
        if (row0 + 8 * hs + r < n) op[r * CH] = acc[r];
    }
  }
}

// Per-node attention logits a_s/a_d = <xp[i], att_*>; also reset m/denom.
__global__ __launch_bounds__(256)
void k_logits(const float* __restrict__ xp,
              const float* __restrict__ att_s, const float* __restrict__ att_d,
              float* __restrict__ a_s, float* __restrict__ a_d,
              unsigned* __restrict__ mkey, float* __restrict__ denom, int n) {
  const int node = blockIdx.x * 8 + (threadIdx.x >> 5);
  const int lane = threadIdx.x & 31;
  if (node >= n) return;
  const float* r = xp + (size_t)node * CH;
  float s = 0.f, d = 0.f;
#pragma unroll
  for (int j = 0; j < 4; ++j) {
    const int c = lane + 32 * j;
    const float v = r[c];
    s += v * att_s[c];
    d += v * att_d[c];
  }
#pragma unroll
  for (int off = 16; off > 0; off >>= 1) {
    s += __shfl_xor(s, off, 32);
    d += __shfl_xor(d, off, 32);
  }
  if (lane == 0) {
    a_s[node] = s;
    a_d[node] = d;
    mkey[node] = 0u;       // < enc of any finite float; every node has a self-loop
    denom[node] = 0.f;
  }
}

// alpha = leaky_relu(a_s[src] + a_d[dst]); segment max via u32 atomic max.
// Edge e >= ne is the self-loop for node (e - ne).
__global__ void k_alpha_max(const int* __restrict__ esrc, const int* __restrict__ edst,
                            const float* __restrict__ a_s, const float* __restrict__ a_d,
                            float* __restrict__ alpha, unsigned* __restrict__ mkey,
                            int ne, int n) {
  int e = blockIdx.x * blockDim.x + threadIdx.x;
  if (e >= ne + n) return;
  int s = (e < ne) ? esrc[e] : (e - ne);
  int t = (e < ne) ? edst[e] : (e - ne);
  float a = a_s[s] + a_d[t];
  a = (a > 0.f) ? a : NEG_SLOPE * a;
  alpha[e] = a;
  atomicMax(&mkey[t], enc_f32(a));
}

// ea = exp(alpha - m[dst]); denom[dst] += ea  (overwrites alpha buffer in place)
__global__ void k_ea_sum(const int* __restrict__ edst,
                         float* __restrict__ alpha_ea,
                         const unsigned* __restrict__ mkey,
                         float* __restrict__ denom, int ne, int n) {
  int e = blockIdx.x * blockDim.x + threadIdx.x;
  if (e >= ne + n) return;
  int t = (e < ne) ? edst[e] : (e - ne);
  float v = __expf(alpha_ea[e] - dec_f32(mkey[t]));
  alpha_ea[e] = v;
  atomic_add_f32(&denom[t], v);
}

// out[dst] += scale * (ea/denom[dst]) * xp[src]; wave per edge, lane = 4 chans.
__global__ __launch_bounds__(256)
void k_scatter(const int* __restrict__ esrc, const int* __restrict__ edst,
               const float* __restrict__ ea, const float* __restrict__ denom,
               const float* __restrict__ xp, float* __restrict__ out,
               float scale, int ne, int n) {
  const int e = blockIdx.x * 8 + (threadIdx.x >> 5);
  const int lane = threadIdx.x & 31;
  if (e >= ne + n) return;
  const int s = (e < ne) ? esrc[e] : (e - ne);
  const int t = (e < ne) ? edst[e] : (e - ne);
  const float coef = scale * ea[e] / (denom[t] + 1e-16f);
  const float4 v = *(const float4*)(xp + (size_t)s * CH + lane * 4);
  float* o = out + (size_t)t * CH + lane * 4;
  atomic_add_f32(o + 0, coef * v.x);
  atomic_add_f32(o + 1, coef * v.y);
  atomic_add_f32(o + 2, coef * v.z);
  atomic_add_f32(o + 3, coef * v.w);
}

// ---------------------------------------------------------------------------

extern "C" void kernel_launch(void* const* d_in, const int* in_sizes, int n_in,
                              void* d_out, int out_size, void* d_ws, size_t ws_size,
                              hipStream_t stream) {
  const int n = in_sizes[0] / CH;            // 50000 nodes
  const float* x = (const float*)d_in[0];
  float* out = (float*)d_out;

  // max edges per hop (all hops equal here, but be general)
  int max_ne = 0;
  for (int h = 0; h < 3; ++h) {
    int ne = in_sizes[1 + 5 * h] / 2;
    if (ne > max_ne) max_ne = ne;
  }

  // workspace carve (256B aligned)
  char* ws = (char*)d_ws;
  size_t off = 0;
  auto carve = [&](size_t bytes) -> char* {
    char* p = ws + off;
    off += (bytes + 255) & ~(size_t)255;
    return p;
  };
  _Float16* x16  = (_Float16*)carve((size_t)n * CH * 2);
  _Float16* wt16 = (_Float16*)carve((size_t)CH * CH * 2);
  float*    xp   = (float*)   carve((size_t)n * CH * 4);
  float*    a_s  = (float*)   carve((size_t)n * 4);
  float*    a_d  = (float*)   carve((size_t)n * 4);
  float*    den  = (float*)   carve((size_t)n * 4);
  unsigned* mkey = (unsigned*)carve((size_t)n * 4);
  float*    eaw  = (float*)   carve((size_t)(max_ne + n) * 4);
  (void)ws_size;

  const float s0 = 1.0f, s1 = 0.9f / 2.0f, s2 = 0.9f / 3.0f;
  const int total_nc = n * CH;

  k_cvt_x<<<(total_nc + 255) / 256, 256, 0, stream>>>(x, x16, total_nc);
  k_init_out<<<(total_nc + 255) / 256, 256, 0, stream>>>(
      out, (const float*)d_in[5], (const float*)d_in[10], (const float*)d_in[15],
      s0, s1, s2, total_nc);

  for (int h = 0; h < 3; ++h) {
    const int base = 1 + 5 * h;
    const int* edge = (const int*)d_in[base];
    const int ne = in_sizes[base] / 2;
    const int* esrc = edge;
    const int* edst = edge + ne;
    const float* W     = (const float*)d_in[base + 1];
    const float* att_s = (const float*)d_in[base + 2];
    const float* att_d = (const float*)d_in[base + 3];
    const float scale = (h == 0) ? s0 : ((h == 1) ? s1 : s2);

    k_cvt_wt<<<(CH * CH + 255) / 256, 256, 0, stream>>>(W, wt16);
    k_gemm_wmma<<<(n + 63) / 64, 128, 0, stream>>>(x16, wt16, xp, n);
    k_logits<<<(n + 7) / 8, 256, 0, stream>>>(xp, att_s, att_d, a_s, a_d, mkey, den, n);
    const int etot = ne + n;
    k_alpha_max<<<(etot + 255) / 256, 256, 0, stream>>>(esrc, edst, a_s, a_d, eaw, mkey, ne, n);
    k_ea_sum<<<(etot + 255) / 256, 256, 0, stream>>>(edst, eaw, mkey, den, ne, n);
    k_scatter<<<(etot + 7) / 8, 256, 0, stream>>>(esrc, edst, eaw, den, xp, out, scale, ne, n);
  }
}
